// SheafBuilderLowRank_31842887533287
// MI455X (gfx1250) — compile-verified, matrix operand order checked by reading, not run on hardware
//
#include <hip/hip_runtime.h>

// CDNA5 / gfx1250 wave32 WMMA types
typedef __attribute__((ext_vector_type(16))) _Float16 v16h;
typedef __attribute__((ext_vector_type(8)))  float    v8f;

#define HID 64
#define DD  6

// K index inside a 16x32 (MxK) f16 A fragment / 32x16 (KxN) B fragment,
// per CDNA5 ISA 7.12.2 "16-bit A-Matrix 16x32":
//   VGPR v (v<4):  lanes 0-15 K=2v,2v+1 ; lanes 16-31 K=8+2v,8+2v+1
//   VGPR v (v>=4): lanes 0-15 K=16+2(v-4).. ; lanes 16-31 K=24+2(v-4)..
__device__ __forceinline__ int kmap(int e, int hl) {
  int v = e >> 1, p = e & 1;
  return ((v & 4) ? 16 : 0) + (hl ? 8 : 0) + ((v & 3) << 1) + p;
}

__device__ __forceinline__ float sigmoidf_(float x) {
  return 1.0f / (1.0f + __expf(-x));
}

// mean over the D=6 stalk dimension: out[n][c] = mean_j in[(n*6+j)][c]
__global__ void mean_over_d_kernel(const float* __restrict__ in,
                                   float* __restrict__ outp, int n) {
  int i = blockIdx.x * blockDim.x + threadIdx.x;
  if (i >= n * HID) return;
  int nn = i >> 6, c = i & 63;
  const float* p = in + ((size_t)nn * DD) * HID + c;
  float s = 0.0f;
#pragma unroll
  for (int j = 0; j < DD; ++j) s += p[(size_t)j * HID];
  outp[i] = s * (1.0f / 6.0f);
}

// One wave (32 lanes) handles 16 NNZ rows: gather -> layernorm -> f16 WMMA
// GEMM (16x128 @ 128x32) -> sigmoid -> 6x6 low-rank block -> transposed store.
__global__ __launch_bounds__(128) void sheaf_main_kernel(
    const float* __restrict__ xm, const float* __restrict__ em,
    const int* __restrict__ he,
    const float* __restrict__ gamma, const float* __restrict__ beta,
    const float* __restrict__ Wm, const float* __restrict__ bias,
    float* __restrict__ out_i0, float* __restrict__ out_i1,
    float* __restrict__ out_at, int nnz)
{
  __shared__ float sh_h[4][16][132];     // padded: bank-conflict-free column reads
  __shared__ float sh_hs[4][16][33];
  __shared__ float sh_red[4][16][2][2];
  __shared__ float sh_mu[4][16];
  __shared__ float sh_rs[4][16];
  __shared__ float sh_g[128];
  __shared__ float sh_b[128];

  const int tid  = threadIdx.x;
  const int w    = tid >> 5;
  const int lane = tid & 31;
  const int n16  = lane & 15;
  const int hl   = lane >> 4;

  __builtin_prefetch(Wm, 0, 0);  // global_prefetch_b8: warm W into GL2

  if (tid < 128) { sh_g[tid] = gamma[tid]; sh_b[tid] = beta[tid]; }

  const int base = blockIdx.x * 64 + w * 16;

  // Stage h = [xm[row] | em[col]] (16 rows x 128 cols) into LDS.
#pragma unroll 4
  for (int r = 0; r < 16; ++r) {
    int idx = base + r; if (idx > nnz - 1) idx = nnz - 1;
    int nr = he[idx];
    int nc = he[nnz + idx];
    const float* xp = xm + (size_t)nr * HID;
    const float* ep = em + (size_t)nc * HID;
    sh_h[w][r][lane]      = xp[lane];
    sh_h[w][r][32 + lane] = xp[32 + lane];
    sh_h[w][r][64 + lane] = ep[lane];
    sh_h[w][r][96 + lane] = ep[32 + lane];
  }
  __syncthreads();

  // LayerNorm statistics: lane (n16,hl) sums half of row n16.
  {
    float s = 0.0f, s2 = 0.0f;
    const float* hr = &sh_h[w][n16][hl * 64];
#pragma unroll
    for (int k = 0; k < 64; ++k) { float v = hr[k]; s += v; s2 += v * v; }
    sh_red[w][n16][hl][0] = s;
    sh_red[w][n16][hl][1] = s2;
  }
  __syncthreads();
  if (hl == 0) {
    float s   = sh_red[w][n16][0][0] + sh_red[w][n16][1][0];
    float s2  = sh_red[w][n16][0][1] + sh_red[w][n16][1][1];
    float mu  = s * (1.0f / 128.0f);
    float var = s2 * (1.0f / 128.0f) - mu * mu;
    sh_mu[w][n16] = mu;
    sh_rs[w][n16] = rsqrtf(var + 1e-5f);
  }
  __syncthreads();

  const float mu = sh_mu[w][n16];
  const float rs = sh_rs[w][n16];

  // A fragments: normalized h rows, K=128 split into 4 chunks of 32.
  v16h A[4];
#pragma unroll
  for (int c = 0; c < 4; ++c) {
#pragma unroll
    for (int e = 0; e < 16; ++e) {
      int k = 32 * c + kmap(e, hl);
      float v = (sh_h[w][n16][k] - mu) * rs * sh_g[k] + sh_b[k];
      A[c][e] = (_Float16)v;
    }
  }

  // B fragments: W (128 x 30, row-major) padded to N=32, two N-tiles.
  v16h Bf[2][4];
#pragma unroll
  for (int t = 0; t < 2; ++t) {
    int col = 16 * t + n16;
#pragma unroll
    for (int c = 0; c < 4; ++c) {
#pragma unroll
      for (int e = 0; e < 16; ++e) {
        int k = 32 * c + kmap(e, hl);
        Bf[t][c][e] = (col < 30) ? (_Float16)Wm[k * 30 + col] : (_Float16)0.0f;
      }
    }
  }

  // D = A x B : 8 x v_wmma_f32_16x16x32_f16 per wave (EXEC all ones here).
  v8f acc0 = {}; v8f acc1 = {};
#pragma unroll
  for (int c = 0; c < 4; ++c)
    acc0 = __builtin_amdgcn_wmma_f32_16x16x32_f16(false, A[c], false, Bf[0][c],
                                                  (short)0, acc0, false, false);
#pragma unroll
  for (int c = 0; c < 4; ++c)
    acc1 = __builtin_amdgcn_wmma_f32_16x16x32_f16(false, A[c], false, Bf[1][c],
                                                  (short)0, acc1, false, false);

  const float b0 = bias[n16];
  const float b1 = (16 + n16 < 30) ? bias[16 + n16] : 0.0f;

  // C/D layout: VGPR i -> M = i (lanes 0-15) / M = 8+i (lanes 16-31), N = lane%16.
#pragma unroll
  for (int i = 0; i < 8; ++i) {
    sh_hs[w][i + 8 * hl][n16]      = sigmoidf_(acc0[i] + b0);
    sh_hs[w][i + 8 * hl][16 + n16] = sigmoidf_(acc1[i] + b1);
  }
  __syncthreads();

  // Epilogue: M[e][f] = sum_r A6[e][r]*B6[f][r] + (e==f)*C6[e]; 36 k's,
  // lanes cover (k = 2*it + hl, row = n16). Transposed layout out[k*nnz + n].
  int midx = base + n16;
  int gi = midx > nnz - 1 ? nnz - 1 : midx;
  int rr = he[gi];
  int cc = he[nnz + gi];
  const bool wr = (midx < nnz);
  const float* hsr = sh_hs[w][n16];
  const float f0r = (float)(6 * rr);
  const float f0c = (float)(6 * cc);

#pragma unroll
  for (int it = 0; it < 18; ++it) {
    int k  = it * 2 + hl;
    int ei = k / 6, fi = k % 6;
    float v = hsr[2 * ei] * hsr[12 + 2 * fi] + hsr[2 * ei + 1] * hsr[13 + 2 * fi];
    if (ei == fi) v += hsr[24 + ei];
    if (wr) {
      size_t o = (size_t)k * (size_t)nnz + (size_t)midx;
      out_i0[o] = f0r + (float)ei;   // 6*row + e  (exact in f32, < 2^24)
      out_i1[o] = f0c + (float)fi;   // 6*col + f
      out_at[o] = v;
    }
  }
}

extern "C" void kernel_launch(void* const* d_in, const int* in_sizes, int n_in,
                              void* d_out, int out_size, void* d_ws, size_t ws_size,
                              hipStream_t stream) {
  const float* x     = (const float*)d_in[0];
  const float* e     = (const float*)d_in[1];
  const int*   he    = (const int*)d_in[2];   // (2, NNZ) int32 (JAX default x64 off)
  const float* gamma = (const float*)d_in[5];
  const float* beta  = (const float*)d_in[6];
  const float* Wm    = (const float*)d_in[7];
  const float* bias  = (const float*)d_in[8];

  const int n_nodes = in_sizes[3];      // 50000
  const int n_edges = in_sizes[4];      // 100000
  const int nnz     = in_sizes[2] / 2;  // 800000

  // workspace: xm (n_nodes*64 f32) + em (n_edges*64 f32) = 38.4 MB
  float* xm = (float*)d_ws;
  float* em = xm + (size_t)n_nodes * HID;

  float* out    = (float*)d_out;
  float* out_i0 = out;                          // sheaf_index[0] as f32
  float* out_i1 = out + (size_t)36 * nnz;       // sheaf_index[1] as f32
  float* out_at = out + (size_t)72 * nnz;       // sheaf_attrs

  const int tx = 256;
  mean_over_d_kernel<<<(n_nodes * HID + tx - 1) / tx, tx, 0, stream>>>(x, xm, n_nodes);
  mean_over_d_kernel<<<(n_edges * HID + tx - 1) / tx, tx, 0, stream>>>(e, em, n_edges);

  const int blocks = (nnz + 63) / 64;   // 64 rows / block (4 waves x 16 rows)
  sheaf_main_kernel<<<blocks, 128, 0, stream>>>(xm, em, he, gamma, beta, Wm, bias,
                                                out_i0, out_i1, out_at, nnz);
}
